// GCN_32049045962841
// MI455X (gfx1250) — matrix operand order repaired and checked by reading, not compiled, not run on hardware
//
#include <hip/hip_runtime.h>
#include <hip/hip_bf16.h>

// ---------------------------------------------------------------------------
// GCN forward for MI455X (gfx1250): bf16 WMMA GEMMs (async global->LDS A-tile
// staging) + fp32-atomic edge aggregation. Wave32 everywhere.
// ---------------------------------------------------------------------------

typedef unsigned short u16;
typedef __attribute__((ext_vector_type(16))) __bf16 v16bf;
typedef __attribute__((ext_vector_type(8)))  float  v8f;

#define GCN_N    20000
#define GCN_E    320000
#define GCN_G    16
#define GCN_DIN  1280
#define GCN_H    256
#define GCN_C    500
#define GCN_EPS  1e-5f

__device__ __forceinline__ u16 f32_to_bf16_rne(float f) {
    unsigned u = __float_as_uint(f);
    unsigned r = (u + 0x7FFFu + ((u >> 16) & 1u)) >> 16;
    return (u16)r;
}

__device__ __forceinline__ void atomAdd(float* p, float v) {
    __hip_atomic_fetch_add(p, v, __ATOMIC_RELAXED, __HIP_MEMORY_SCOPE_AGENT);
}

// ------------------------------ utility kernels -----------------------------

__global__ void k_zero_f32(float* __restrict__ p, int n) {
    int i = blockIdx.x * blockDim.x + threadIdx.x;
    if (i < n) p[i] = 0.0f;
}

__global__ void k_f32_to_bf16(const float* __restrict__ in, u16* __restrict__ out, int n) {
    int i = blockIdx.x * blockDim.x + threadIdx.x;
    if (i < n) out[i] = f32_to_bf16_rne(in[i]);
}

// ------------------------------ graph kernels -------------------------------

__global__ void k_deg(const int* __restrict__ col, const float* __restrict__ ew,
                      float* __restrict__ deg, int E) {
    int e = blockIdx.x * blockDim.x + threadIdx.x;
    if (e < E) atomAdd(&deg[col[e]], ew[e]);
}

__global__ void k_dinv(const float* __restrict__ deg, float* __restrict__ dinv, int n) {
    int i = blockIdx.x * blockDim.x + threadIdx.x;
    if (i < n) dinv[i] = rsqrtf(deg[i] + 1.0f);
}

__global__ void k_norm(const int* __restrict__ row, const int* __restrict__ col,
                       const float* __restrict__ ew, const float* __restrict__ dinv,
                       float* __restrict__ nrm, int E) {
    int e = blockIdx.x * blockDim.x + threadIdx.x;
    if (e < E) nrm[e] = dinv[row[e]] * ew[e] * dinv[col[e]];
}

// agg[i,:] = dinv[i]^2 * xw[i,:]   (self term; full overwrite, no pre-zero)
__global__ void k_self_init(const float* __restrict__ dinv, const float* __restrict__ xw,
                            float* __restrict__ agg, int total) {
    int idx = blockIdx.x * blockDim.x + threadIdx.x;
    if (idx < total) {
        int i = idx >> 8;               // H == 256
        float d = dinv[i];
        agg[idx] = d * d * xw[idx];
    }
}

// one wave per edge: agg[col,:] += norm[e] * xw[row,:]
// edge scalars made wave-uniform (readfirstlane) -> scalar loads
__global__ __launch_bounds__(256)
void k_edge_agg(const int* __restrict__ row, const int* __restrict__ col,
                const float* __restrict__ nrm, const float* __restrict__ xw,
                float* __restrict__ agg, int E) {
    int wid  = (int)((blockIdx.x * 256u + threadIdx.x) >> 5);
    wid = __builtin_amdgcn_readfirstlane(wid);
    int lane = threadIdx.x & 31;
    if (wid >= E) return;
    int r = __builtin_amdgcn_readfirstlane(row[wid]);
    int c = __builtin_amdgcn_readfirstlane(col[wid]);
    float nv = nrm[wid];
    const float* src = xw + (size_t)r * GCN_H;
    float*       dst = agg + (size_t)c * GCN_H;
#pragma unroll
    for (int j = 0; j < 8; ++j) {
        int k = lane + j * 32;
        atomAdd(&dst[k], nv * src[k]);
    }
}

// ------------------------------ batch norm ----------------------------------

// coalesced: block b sweeps rows b, b+grid, ...; lane c owns column c
__global__ __launch_bounds__(256)
void k_bn_stats(const float* __restrict__ agg, const float* __restrict__ bias,
                float* __restrict__ ssum, float* __restrict__ ssq,
                int n_rows, int relu) {
    int c = threadIdx.x;
    float b = bias[c];
    float s = 0.0f, s2 = 0.0f;
    for (int i = blockIdx.x; i < n_rows; i += gridDim.x) {
        float v = agg[(size_t)i * GCN_H + c] + b;
        if (relu) v = fmaxf(v, 0.0f);
        s += v; s2 += v * v;
    }
    atomAdd(&ssum[c], s);
    atomAdd(&ssq[c], s2);
}

__global__ void k_bn_finalize(const float* __restrict__ ssum, const float* __restrict__ ssq,
                              float* __restrict__ mean, float* __restrict__ rstd, float inv_n) {
    int c = threadIdx.x;
    float m = ssum[c] * inv_n;
    mean[c] = m;
    float v = ssq[c] * inv_n - m * m;
    rstd[c] = rsqrtf(v + GCN_EPS);
}

__global__ void k_bn_apply(const float* __restrict__ agg, const float* __restrict__ bias,
                           const float* __restrict__ mean, const float* __restrict__ rstd,
                           const float* __restrict__ g, const float* __restrict__ be,
                           u16* __restrict__ hbf, float* __restrict__ hf32,
                           int total, int relu) {
    int idx = blockIdx.x * blockDim.x + threadIdx.x;
    if (idx >= total) return;
    int c = idx & 255;
    float v = agg[idx] + bias[c];
    if (relu) v = fmaxf(v, 0.0f);
    v = (v - mean[c]) * rstd[c] * g[c] + be[c];
    hbf[idx] = f32_to_bf16_rne(v);
    if (hf32) hf32[idx] = v;
}

// ------------------------------ pooling / combine ---------------------------

__global__ void k_pool(const float* __restrict__ h, const int* __restrict__ batch,
                       float* __restrict__ pooled, float* __restrict__ cnt, int total) {
    int idx = blockIdx.x * blockDim.x + threadIdx.x;
    if (idx >= total) return;
    int i = idx >> 8;
    int c = idx & 255;
    int g = batch[i];
    atomAdd(&pooled[g * GCN_H + c], h[idx]);
    if (c == 0) atomAdd(&cnt[g], 1.0f);
}

__global__ void k_combine(const float* __restrict__ pooled, const float* __restrict__ cnt,
                          const float* __restrict__ y, u16* __restrict__ zbf, int total) {
    int idx = blockIdx.x * blockDim.x + threadIdx.x;
    if (idx >= total) return;
    int g = idx >> 8;
    float v = pooled[idx] / fmaxf(cnt[g], 1.0f) + y[idx];
    zbf[idx] = f32_to_bf16_rne(v);
}

// ------------------------------ WMMA GEMM -----------------------------------
// C[M,N] (f32) = A[M,K](bf16,row-major) @ B[K,N](bf16,row-major) (+bias,+sigmoid)
// Block: 256 threads = 8 waves, 128x64 tile, BK=32.
// Wave w: rows [ (w&3)*32, +32 ), cols [ (w>>2)*32, +32 ) -> four 16x16 accums.
// A tile staged with GLOBAL_LOAD_ASYNC_TO_LDS_B128 (ASYNCcnt) on full blocks.

#define BM 128
#define BN 64
#define BK 32
#define BKP 40   // padded LDS K-stride (elements); 80B rows keep b128 alignment

union Frag16 { uint4 u[2]; v16bf v; };
union FragC  { float f[8]; v8f v; };

__global__ __launch_bounds__(256)
void k_gemm_bf16_wmma(const u16* __restrict__ A, const u16* __restrict__ B,
                      const float* __restrict__ bias, float* __restrict__ C,
                      int M, int N, int K, int fuse_sigmoid) {
    __shared__ u16 ldsA[BM * BKP];   // [m][k]
    __shared__ u16 ldsB[BN * BKP];   // transposed: [n][k]

    const int tid  = threadIdx.x;
    const int lane = tid & 31;
    const int wave = tid >> 5;
    const int block_m = blockIdx.x * BM;
    const int block_n = blockIdx.y * BN;

    const int wm   = wave & 3;       // 32-row stripe within block
    const int wn   = wave >> 2;      // 0/1 -> 32-col stripe
    const int half = lane >> 4;      // K-half selector per ISA layout
    const int l16  = lane & 15;

    v8f c00 = {}, c01 = {}, c10 = {}, c11 = {};

    // global->LDS assignments
    const int arow = tid >> 1;            // 0..127
    const int aseg = (tid & 1) * 16;      // 0 or 16 (bf16 elements)
    const int bk   = tid >> 3;            // 0..31
    const int bseg = (tid & 7) * 8;       // 0..56
    const bool n_full = (block_n + BN) <= N;   // block-uniform
    const bool m_full = (block_m + BM) <= M;   // block-uniform

    // LDS byte offset of this thread's A staging slot (low 32 bits of the
    // generic pointer are the LDS offset per the flat-aperture mapping)
    const unsigned ldsA_off = (unsigned)(uintptr_t)(&ldsA[arow * BKP + aseg]);

    for (int k0 = 0; k0 < K; k0 += BK) {
        __syncthreads();
        // A tile: 128 rows x 32 K, 32 bytes per thread
        {
            int gr = block_m + arow;
            const u16* ap = A + (size_t)gr * K + k0 + aseg;
            if (m_full) {
                // async copy straight to LDS, no VGPR round-trip (ASYNCcnt)
                unsigned long long ga = (unsigned long long)(uintptr_t)ap;
                asm volatile(
                    "global_load_async_to_lds_b128 %0, %1, off\n\t"
                    "global_load_async_to_lds_b128 %2, %3, off"
                    :: "v"(ldsA_off), "v"(ga),
                       "v"(ldsA_off + 16u), "v"(ga + 16ull)
                    : "memory");
                if (k0 + BK < K) __builtin_prefetch(ap + BK, 0, 1); // global_prefetch
            } else {
                uint4 av0 = {0u, 0u, 0u, 0u}, av1 = {0u, 0u, 0u, 0u};
                if (gr < M) {
                    av0 = *(const uint4*)ap;
                    av1 = *(const uint4*)(ap + 8);
                }
                *(uint4*)(&ldsA[arow * BKP + aseg])     = av0;
                *(uint4*)(&ldsA[arow * BKP + aseg + 8]) = av1;
            }
        }
        // B tile: 32 K x 64 N, stored transposed [n][k]
        {
            int gk = k0 + bk;
            const u16* bp = B + (size_t)gk * N + block_n + bseg;
            if (n_full) {
                // branch-free fast path: one b128 load + 8 ds stores
                uint4 bv = *(const uint4*)bp;
                const u16* be = (const u16*)&bv;
#pragma unroll
                for (int j = 0; j < 8; ++j)
                    ldsB[(bseg + j) * BKP + bk] = be[j];
            } else {
#pragma unroll
                for (int j = 0; j < 8; ++j) {
                    int gn = block_n + bseg + j;
                    u16 bv = (gn < N) ? bp[j] : (u16)0;
                    ldsB[(bseg + j) * BKP + bk] = bv;
                }
            }
        }
        // wait for this wave's async LDS writes, then block-wide barrier
        asm volatile("s_wait_asynccnt 0x0" ::: "memory");
        __syncthreads();

        // A fragments (two 16-row tiles): ISA 16-bit A 16x32 layout
        Frag16 a0, a1, b0, b1;
        const int am0   = wm * 32 + l16;
        const int am1   = am0 + 16;
        const int akoff = half * 8;
        a0.u[0] = *(const uint4*)(&ldsA[am0 * BKP + akoff]);
        a0.u[1] = *(const uint4*)(&ldsA[am0 * BKP + 16 + akoff]);
        a1.u[0] = *(const uint4*)(&ldsA[am1 * BKP + akoff]);
        a1.u[1] = *(const uint4*)(&ldsA[am1 * BKP + 16 + akoff]);

        // B fragments: lanes 0-15 K=0..7,16..23, lanes 16-31 K=8..15,24..31, col = l16
        const int bkoff = half * 8;
        const int bn0 = wn * 32 + l16;
        const int bn1 = bn0 + 16;
        b0.u[0] = *(const uint4*)(&ldsB[bn0 * BKP + bkoff]);
        b0.u[1] = *(const uint4*)(&ldsB[bn0 * BKP + bkoff + 16]);
        b1.u[0] = *(const uint4*)(&ldsB[bn1 * BKP + bkoff]);
        b1.u[1] = *(const uint4*)(&ldsB[bn1 * BKP + bkoff + 16]);

        c00 = __builtin_amdgcn_wmma_f32_16x16x32_bf16(false, a0.v, false, b0.v,
                                                      (short)0, c00, false, false);
        c01 = __builtin_amdgcn_wmma_f32_16x16x32_bf16(false, a0.v, false, b1.v,
                                                      (short)0, c01, false, false);
        c10 = __builtin_amdgcn_wmma_f32_16x16x32_bf16(false, a1.v, false, b0.v,
                                                      (short)0, c10, false, false);
        c11 = __builtin_amdgcn_wmma_f32_16x16x32_bf16(false, a1.v, false, b1.v,
                                                      (short)0, c11, false, false);
    }

    // C/D layout: col = l16, rows = half*8 + r
    const int ccol0 = block_n + wn * 32 + l16;
    const int ccol1 = ccol0 + 16;
    const float bias0 = (bias && ccol0 < N) ? bias[ccol0] : 0.0f;
    const float bias1 = (bias && ccol1 < N) ? bias[ccol1] : 0.0f;

    FragC fc[4];
    fc[0].v = c00; fc[1].v = c01; fc[2].v = c10; fc[3].v = c11;
#pragma unroll
    for (int t = 0; t < 4; ++t) {
        const int mt   = t >> 1;                 // A tile index
        const int nt   = t & 1;                  // B tile index
        const int crow = block_m + wm * 32 + mt * 16 + half * 8;
        const int ccol = nt ? ccol1 : ccol0;
        const float bb = nt ? bias1 : bias0;
        if (ccol >= N) continue;
#pragma unroll
        for (int r = 0; r < 8; ++r) {
            int gr = crow + r;
            if (gr >= M) continue;
            float v = fc[t].f[r] + bb;
            if (fuse_sigmoid) v = 1.0f / (1.0f + __expf(-v));
            C[(size_t)gr * N + ccol] = v;
        }
    }
}

// ------------------------------ host orchestration --------------------------

extern "C" void kernel_launch(void* const* d_in, const int* in_sizes, int n_in,
                              void* d_out, int out_size, void* d_ws, size_t ws_size,
                              hipStream_t stream) {
    (void)in_sizes; (void)n_in; (void)out_size; (void)ws_size;

    const float* x     = (const float*)d_in[0];
    const int*   erow  = (const int*)d_in[1];
    const int*   ecol  = (const int*)d_in[1] + GCN_E;
    const float* ea    = (const float*)d_in[2];
    const int*   batch = (const int*)d_in[3];
    const float* seq   = (const float*)d_in[4];
    const float* W1 = (const float*)d_in[5],  *b1 = (const float*)d_in[6];
    const float* g1 = (const float*)d_in[7],  *be1 = (const float*)d_in[8];
    const float* W2 = (const float*)d_in[9],  *b2 = (const float*)d_in[10];
    const float* g2 = (const float*)d_in[11], *be2 = (const float*)d_in[12];
    const float* W3 = (const float*)d_in[13], *b3 = (const float*)d_in[14];
    const float* g3 = (const float*)d_in[15], *be3 = (const float*)d_in[16];
    const float* fc1_w = (const float*)d_in[17], *fc1_b = (const float*)d_in[18];
    const float* lin_w = (const float*)d_in[19], *lin_b = (const float*)d_in[20];
    float* out = (float*)d_out;

    // workspace layout
    char* ws = (char*)d_ws;
    size_t off = 0;
    auto alloc = [&](size_t bytes) -> void* {
        off = (off + 255) & ~(size_t)255;
        void* p = ws + off;
        off += bytes;
        return p;
    };
    u16* xbf    = (u16*)alloc((size_t)GCN_N * GCN_DIN * 2);
    u16* w1bf   = (u16*)alloc((size_t)GCN_DIN * GCN_H * 2);
    u16* w2bf   = (u16*)alloc((size_t)GCN_H * GCN_H * 2);
    u16* w3bf   = (u16*)alloc((size_t)GCN_H * GCN_H * 2);
    u16* fc1bf  = (u16*)alloc((size_t)GCN_DIN * GCN_H * 2);
    u16* linbf  = (u16*)alloc((size_t)GCN_H * GCN_C * 2);
    u16* seqbf  = (u16*)alloc((size_t)GCN_G * GCN_DIN * 2);
    u16* hbf    = (u16*)alloc((size_t)GCN_N * GCN_H * 2);
    u16* zbf    = (u16*)alloc((size_t)GCN_G * GCN_H * 2);
    float* xw     = (float*)alloc((size_t)GCN_N * GCN_H * 4);
    float* agg    = (float*)alloc((size_t)GCN_N * GCN_H * 4);
    float* hf32   = (float*)alloc((size_t)GCN_N * GCN_H * 4);
    float* deg    = (float*)alloc((size_t)GCN_N * 4);
    float* dinv   = (float*)alloc((size_t)GCN_N * 4);
    float* nrm    = (float*)alloc((size_t)GCN_E * 4);
    float* ssum   = (float*)alloc((size_t)2 * GCN_H * 4);   // ssum | ssq
    float* ssq    = ssum + GCN_H;
    float* meanv  = (float*)alloc((size_t)GCN_H * 4);
    float* rstdv  = (float*)alloc((size_t)GCN_H * 4);
    float* pooled = (float*)alloc((size_t)GCN_G * GCN_H * 4);
    float* cnt    = (float*)alloc((size_t)GCN_G * 4);
    float* ybuf   = (float*)alloc((size_t)GCN_G * GCN_H * 4);

    const int T = 256;
    auto cdiv = [](long a, long b) { return (int)((a + b - 1) / b); };

    // precision conversions (bf16 operands for WMMA)
    k_f32_to_bf16<<<cdiv((long)GCN_N * GCN_DIN, T), T, 0, stream>>>(x, xbf, GCN_N * GCN_DIN);
    k_f32_to_bf16<<<cdiv((long)GCN_DIN * GCN_H, T), T, 0, stream>>>(W1, w1bf, GCN_DIN * GCN_H);
    k_f32_to_bf16<<<cdiv((long)GCN_H * GCN_H, T), T, 0, stream>>>(W2, w2bf, GCN_H * GCN_H);
    k_f32_to_bf16<<<cdiv((long)GCN_H * GCN_H, T), T, 0, stream>>>(W3, w3bf, GCN_H * GCN_H);
    k_f32_to_bf16<<<cdiv((long)GCN_DIN * GCN_H, T), T, 0, stream>>>(fc1_w, fc1bf, GCN_DIN * GCN_H);
    k_f32_to_bf16<<<cdiv((long)GCN_H * GCN_C, T), T, 0, stream>>>(lin_w, linbf, GCN_H * GCN_C);
    k_f32_to_bf16<<<cdiv((long)GCN_G * GCN_DIN, T), T, 0, stream>>>(seq, seqbf, GCN_G * GCN_DIN);

    // degree / normalization (shared by all layers)
    k_zero_f32<<<cdiv(GCN_N, T), T, 0, stream>>>(deg, GCN_N);
    k_deg<<<cdiv(GCN_E, T), T, 0, stream>>>(ecol, ea, deg, GCN_E);
    k_dinv<<<cdiv(GCN_N, T), T, 0, stream>>>(deg, dinv, GCN_N);
    k_norm<<<cdiv(GCN_E, T), T, 0, stream>>>(erow, ecol, ea, dinv, nrm, GCN_E);

    dim3 gemm_grid(cdiv(GCN_N, BM), cdiv(GCN_H, BN));
    const int NH = GCN_N * GCN_H;

    auto run_layer = [&](const u16* inbf, int Kdim, const u16* wbf, const float* bb,
                         const float* gg, const float* bbeta, int relu,
                         u16* houtbf, float* houtf) {
        k_gemm_bf16_wmma<<<gemm_grid, 256, 0, stream>>>(inbf, wbf, nullptr, xw,
                                                        GCN_N, GCN_H, Kdim, 0);
        k_self_init<<<cdiv(NH, T), T, 0, stream>>>(dinv, xw, agg, NH);
        k_edge_agg<<<cdiv((long)GCN_E * 32, T), T, 0, stream>>>(erow, ecol, nrm, xw, agg, GCN_E);
        k_zero_f32<<<cdiv(2 * GCN_H, T), T, 0, stream>>>(ssum, 2 * GCN_H);
        k_bn_stats<<<256, 256, 0, stream>>>(agg, bb, ssum, ssq, GCN_N, relu);
        k_bn_finalize<<<1, GCN_H, 0, stream>>>(ssum, ssq, meanv, rstdv, 1.0f / GCN_N);
        k_bn_apply<<<cdiv(NH, T), T, 0, stream>>>(agg, bb, meanv, rstdv, gg, bbeta,
                                                  houtbf, houtf, NH, relu);
    };

    run_layer(xbf, GCN_DIN, w1bf, b1, g1, be1, 1, hbf, nullptr);
    run_layer(hbf, GCN_H,   w2bf, b2, g2, be2, 1, hbf, nullptr);
    run_layer(hbf, GCN_H,   w3bf, b3, g3, be3, 0, hbf, hf32);

    // mean-pool per graph
    k_zero_f32<<<cdiv(GCN_G * GCN_H, T), T, 0, stream>>>(pooled, GCN_G * GCN_H);
    k_zero_f32<<<1, T, 0, stream>>>(cnt, GCN_G);
    k_pool<<<cdiv(NH, T), T, 0, stream>>>(hf32, batch, pooled, cnt, NH);

    // y = seq_emb @ fc1_w + fc1_b   (16 x 1280 x 256)
    k_gemm_bf16_wmma<<<dim3(1, cdiv(GCN_H, BN)), 256, 0, stream>>>(
        seqbf, fc1bf, fc1_b, ybuf, GCN_G, GCN_H, GCN_DIN, 0);

    // z = pooled/cnt + y  (bf16 for final GEMM)
    k_combine<<<cdiv(GCN_G * GCN_H, T), T, 0, stream>>>(pooled, cnt, ybuf, zbf, GCN_G * GCN_H);

    // out = sigmoid(z @ lin_w + lin_b)   (16 x 256 x 500)
    k_gemm_bf16_wmma<<<dim3(1, cdiv(GCN_C, BN)), 256, 0, stream>>>(
        zbf, linbf, lin_b, out, GCN_G, GCN_C, GCN_H, 1);
}
